// GRU_RNN_85633057947864
// MI455X (gfx1250) — matrix operand order repaired and checked by reading, not compiled, not run on hardware
//
#include <hip/hip_runtime.h>
#include <hip/hip_bf16.h>
#include <stdint.h>

// GRU RNN on gfx1250: bf16 WMMA (v_wmma_f32_16x16x32_bf16) recurrence.
// B=64, I=512, H=1024, T=512.

typedef __attribute__((ext_vector_type(16))) __bf16 v16bf;
typedef __attribute__((ext_vector_type(8)))  float  v8f;
typedef int v4i __attribute__((vector_size(16)));
typedef __attribute__((address_space(1))) v4i* gbl_v4i;
typedef __attribute__((address_space(3))) v4i* lds_v4i;

#define B_  64
#define I_  512
#define H_  1024
#define G3_ 3072   // 3*H
#define T_  512

#if defined(__has_builtin)
#if __has_builtin(__builtin_amdgcn_global_load_async_to_lds_b128) && \
    __has_builtin(__builtin_amdgcn_s_wait_asynccnt)
#define USE_ASYNC_LDS 1
#endif
#endif

__device__ __forceinline__ unsigned short f32_to_bf16(float f) {
    union { float f; uint32_t u; } x; x.f = f;
    uint32_t u = x.u;
    u += 0x7FFFu + ((u >> 16) & 1u);      // round-to-nearest-even
    return (unsigned short)(u >> 16);
}
__device__ __forceinline__ float bf16_to_f32(unsigned short h) {
    union { uint32_t u; float f; } x; x.u = ((uint32_t)h) << 16;
    return x.f;
}

struct alignas(16) H8 { unsigned short h[8]; };   // 16-byte chunk of 8 bf16
union ABF { v16bf v; H8 half[2]; };

// ---------------- phase 0: conversions ----------------

__global__ void k_f32_to_bf16(const float* __restrict__ src,
                              unsigned short* __restrict__ dst, int n) {
    int i = blockIdx.x * blockDim.x + threadIdx.x;
    if (i < n) dst[i] = f32_to_bf16(src[i]);
}

// inputs [B,I,T] f32  ->  xbf [T,B,I] bf16
__global__ void k_transpose_x(const float* __restrict__ x,
                              unsigned short* __restrict__ xt) {
    int idx = blockIdx.x * blockDim.x + threadIdx.x;   // linear over (b,i,t), t fastest
    if (idx >= B_ * I_ * T_) return;
    int t  = idx % T_;
    int bi = idx / T_;
    int i  = bi % I_;
    int b  = bi / I_;
    xt[((size_t)t * B_ + b) * I_ + i] = f32_to_bf16(x[idx]);
}

__global__ void k_zero_u16(unsigned short* __restrict__ p, int n) {
    int i = blockIdx.x * blockDim.x + threadIdx.x;
    if (i < n) p[i] = 0;
}

// ---------------- phase 1: one GRU time step ----------------
// grid: (8 j-groups, 4 batch-tiles), block: 256 threads = 8 waves.
// Each wave owns 16 hidden columns; computes 6 WMMA accumulators:
// gh_{r,z,n} over K=H and gi_{r,z,n} over K=I for its (Mtile=16, Ntile=16).

__global__ __launch_bounds__(256)
void k_gru_step(const unsigned short* __restrict__ hprev,  // [B,H] bf16
                unsigned short*       __restrict__ hnext,  // [B,H] bf16
                const unsigned short* __restrict__ xbf,    // [T,B,I] bf16
                const unsigned short* __restrict__ wih,    // [3H,I]  bf16
                const unsigned short* __restrict__ whh,    // [3H,H]  bf16
                const float* __restrict__ bih,             // [3H]
                const float* __restrict__ bhh,             // [3H]
                float* __restrict__ allh,                  // [B,H,T] f32 out
                int t)
{
    __shared__ unsigned short hA[16 * H_];   // 32 KB: h tile, A-operand source
    __shared__ unsigned short xA[16 * I_];   // 16 KB: x tile

    const int b_base = blockIdx.y * 16;
    const int tid    = threadIdx.x;

    // Stage the (contiguous) h and x tiles into LDS.
    {
        const uint4* hsrc = reinterpret_cast<const uint4*>(hprev + (size_t)b_base * H_);
        uint4*       hdst = reinterpret_cast<uint4*>(hA);
        const uint4* xsrc = reinterpret_cast<const uint4*>(xbf + ((size_t)t * B_ + b_base) * I_);
        uint4*       xdst = reinterpret_cast<uint4*>(xA);
#ifdef USE_ASYNC_LDS
        // gfx1250 direct global->LDS async DMA (ASYNCcnt), no VGPR bounce.
        for (int i = tid; i < 16 * H_ / 8; i += 256)
            __builtin_amdgcn_global_load_async_to_lds_b128(
                (gbl_v4i)(hsrc + i), (lds_v4i)(hdst + i), 0, 0);
        for (int i = tid; i < 16 * I_ / 8; i += 256)
            __builtin_amdgcn_global_load_async_to_lds_b128(
                (gbl_v4i)(xsrc + i), (lds_v4i)(xdst + i), 0, 0);
        __builtin_amdgcn_s_wait_asynccnt(0);
#else
        for (int i = tid; i < 16 * H_ / 8; i += 256) hdst[i] = hsrc[i];
        for (int i = tid; i < 16 * I_ / 8; i += 256) xdst[i] = xsrc[i];
#endif
    }
    __syncthreads();

    const int wave  = tid >> 5;
    const int lane  = tid & 31;
    const int ln    = lane & 15;          // N (and A-row M index for operand fetch)
    const int hi    = lane >> 4;          // half-wave select
    const int j_base = (blockIdx.x * 8 + wave) * 16;   // hidden-column tile

    // Weight rows for the three gates of column (j_base + ln).
    const int g0 = 0 * H_ + j_base + ln;  // r gate
    const int g1 = 1 * H_ + j_base + ln;  // z gate
    const int g2 = 2 * H_ + j_base + ln;  // n gate

    v8f ah0 = {}, ah1 = {}, ah2 = {};     // gh accumulators
    v8f ax0 = {}, ax1 = {}, ax2 = {};     // gi accumulators

    // ---- gh = h · W_hh^T  (K = 1024) ----
    for (int k0 = 0; k0 < H_; k0 += 32) {
        ABF a;
        // A (16x32 bf16): lanes 0-15 K=0..15, lanes 16-31 K=16..31
        a.half[0] = *reinterpret_cast<const H8*>(&hA[ln * H_ + k0 + hi * 16]);
        a.half[1] = *reinterpret_cast<const H8*>(&hA[ln * H_ + k0 + hi * 16 + 8]);
        // B (32x16 bf16): lane = column, lanes 0-15 K=0..15, lanes 16-31 K=16..31
        ABF b0, b1, b2;
        const unsigned short* w0 = whh + (size_t)g0 * H_ + k0 + hi * 16;
        const unsigned short* w1 = whh + (size_t)g1 * H_ + k0 + hi * 16;
        const unsigned short* w2 = whh + (size_t)g2 * H_ + k0 + hi * 16;
        b0.half[0] = *reinterpret_cast<const H8*>(w0);
        b0.half[1] = *reinterpret_cast<const H8*>(w0 + 8);
        b1.half[0] = *reinterpret_cast<const H8*>(w1);
        b1.half[1] = *reinterpret_cast<const H8*>(w1 + 8);
        b2.half[0] = *reinterpret_cast<const H8*>(w2);
        b2.half[1] = *reinterpret_cast<const H8*>(w2 + 8);
        ah0 = __builtin_amdgcn_wmma_f32_16x16x32_bf16(false, a.v, false, b0.v, (short)0, ah0, false, false);
        ah1 = __builtin_amdgcn_wmma_f32_16x16x32_bf16(false, a.v, false, b1.v, (short)0, ah1, false, false);
        ah2 = __builtin_amdgcn_wmma_f32_16x16x32_bf16(false, a.v, false, b2.v, (short)0, ah2, false, false);
    }

    // ---- gi = x · W_ih^T  (K = 512) ----
    for (int k0 = 0; k0 < I_; k0 += 32) {
        ABF a;
        a.half[0] = *reinterpret_cast<const H8*>(&xA[ln * I_ + k0 + hi * 16]);
        a.half[1] = *reinterpret_cast<const H8*>(&xA[ln * I_ + k0 + hi * 16 + 8]);
        ABF b0, b1, b2;
        const unsigned short* w0 = wih + (size_t)g0 * I_ + k0 + hi * 16;
        const unsigned short* w1 = wih + (size_t)g1 * I_ + k0 + hi * 16;
        const unsigned short* w2 = wih + (size_t)g2 * I_ + k0 + hi * 16;
        b0.half[0] = *reinterpret_cast<const H8*>(w0);
        b0.half[1] = *reinterpret_cast<const H8*>(w0 + 8);
        b1.half[0] = *reinterpret_cast<const H8*>(w1);
        b1.half[1] = *reinterpret_cast<const H8*>(w1 + 8);
        b2.half[0] = *reinterpret_cast<const H8*>(w2);
        b2.half[1] = *reinterpret_cast<const H8*>(w2 + 8);
        ax0 = __builtin_amdgcn_wmma_f32_16x16x32_bf16(false, a.v, false, b0.v, (short)0, ax0, false, false);
        ax1 = __builtin_amdgcn_wmma_f32_16x16x32_bf16(false, a.v, false, b1.v, (short)0, ax1, false, false);
        ax2 = __builtin_amdgcn_wmma_f32_16x16x32_bf16(false, a.v, false, b2.v, (short)0, ax2, false, false);
    }

    // Biases are per gate-column (depend on N = ln only).
    const float bi0 = bih[g0], bi1 = bih[g1], bi2 = bih[g2];
    const float bh0 = bhh[g0], bh1 = bhh[g1], bh2 = bhh[g2];

    // C/D layout: element r, lane -> M = r + 8*hi, N = ln.
    const int j = j_base + ln;
    #pragma unroll
    for (int r = 0; r < 8; ++r) {
        const int m = r + hi * 8;
        float gr = ax0[r] + bi0 + ah0[r] + bh0;
        float gz = ax1[r] + bi1 + ah1[r] + bh1;
        float hn = ah2[r] + bh2;
        float rg = 1.0f / (1.0f + __expf(-gr));
        float zg = 1.0f / (1.0f + __expf(-gz));
        float ng = tanhf(ax2[r] + bi2 + rg * hn);
        float hp = bf16_to_f32(hA[m * H_ + j]);
        float hv = (1.0f - zg) * ng + zg * hp;
        hnext[(size_t)(b_base + m) * H_ + j] = f32_to_bf16(hv);
        // Stream the 134 MB hidden trace non-temporally: keep L2 for weights/h/x.
        __builtin_nontemporal_store(hv, &allh[((size_t)(b_base + m) * H_ + j) * T_ + t]);
    }
}

// ---------------- phase 2: output projection ----------------
// out[b,t] = b_out + sum_j allh[b,j,t] * w_out[j]   (coalesced over t)
__global__ void k_out_proj(const float* __restrict__ allh,
                           const float* __restrict__ wout,
                           const float* __restrict__ bout,
                           float* __restrict__ out) {
    int t = blockIdx.x * blockDim.x + threadIdx.x;
    int b = blockIdx.y;
    if (t >= T_) return;
    float s = bout[0];
    const float* hb = allh + (size_t)b * H_ * T_ + t;
    for (int j = 0; j < H_; ++j)
        s += __builtin_nontemporal_load(&hb[(size_t)j * T_]) * wout[j];
    out[(size_t)b * T_ + t] = s;
}

// ---------------- launch ----------------

extern "C" void kernel_launch(void* const* d_in, const int* in_sizes, int n_in,
                              void* d_out, int out_size, void* d_ws, size_t ws_size,
                              hipStream_t stream) {
    (void)in_sizes; (void)n_in; (void)out_size; (void)ws_size;
    const float* inputs = (const float*)d_in[0];   // [B,I,T]
    const float* w_ih   = (const float*)d_in[1];   // [3H,I]
    const float* w_hh   = (const float*)d_in[2];   // [3H,H]
    const float* b_ih   = (const float*)d_in[3];   // [3H]
    const float* b_hh   = (const float*)d_in[4];   // [3H]
    const float* w_out  = (const float*)d_in[5];   // [1,H]
    const float* b_out  = (const float*)d_in[6];   // [1]

    float* out  = (float*)d_out;                   // [B,T]
    float* allh = out + (size_t)B_ * T_;           // [B,H,T]

    // workspace carve-out
    char* ws = (char*)d_ws;
    size_t off = 0;
    auto carve = [&](size_t bytes) {
        char* p = ws + off;
        off = (off + bytes + 255) & ~(size_t)255;
        return p;
    };
    unsigned short* whh_bf = (unsigned short*)carve((size_t)G3_ * H_ * 2);     // 6 MB
    unsigned short* wih_bf = (unsigned short*)carve((size_t)G3_ * I_ * 2);     // 3 MB
    unsigned short* xbf    = (unsigned short*)carve((size_t)T_ * B_ * I_ * 2); // 33.5 MB
    unsigned short* h0     = (unsigned short*)carve((size_t)B_ * H_ * 2);
    unsigned short* h1     = (unsigned short*)carve((size_t)B_ * H_ * 2);

    int n;
    n = G3_ * H_;
    k_f32_to_bf16<<<(n + 255) / 256, 256, 0, stream>>>(w_hh, whh_bf, n);
    n = G3_ * I_;
    k_f32_to_bf16<<<(n + 255) / 256, 256, 0, stream>>>(w_ih, wih_bf, n);
    n = B_ * I_ * T_;
    k_transpose_x<<<(n + 255) / 256, 256, 0, stream>>>(inputs, xbf);
    n = B_ * H_;
    k_zero_u16<<<(n + 255) / 256, 256, 0, stream>>>(h0, n);

    for (int t = 0; t < T_; ++t) {
        unsigned short* hp = (t & 1) ? h1 : h0;
        unsigned short* hn = (t & 1) ? h0 : h1;
        k_gru_step<<<dim3(8, 4), 256, 0, stream>>>(hp, hn, xbf, wih_bf, whh_bf,
                                                   b_ih, b_hh, allh, t);
    }

    k_out_proj<<<dim3(T_ / 256, B_), 256, 0, stream>>>(allh, w_out, b_out, out);
}
